// Encoder_68977174774136
// MI455X (gfx1250) — matrix operand order repaired and checked by reading, not compiled, not run on hardware
//
#include <hip/hip_runtime.h>
#include <hip/hip_bf16.h>

// ---------------------------------------------------------------------------
// CDNA5 (gfx1250) hetero-GCN encoder.
// bf16 WMMA (v_wmma_f32_16x16x32_bf16) for all GEMMs, f32 accumulation.
// wave32; 256-thread WGs = 8 waves, each wave owns a 16-wide column slice.
// Aggregation: one LDS staging phase (3 edge types x 16 rows x 416 K, bf16)
// per output tile, then 13x3 WMMAs with 3 independent accumulator chains.
// ---------------------------------------------------------------------------

typedef __bf16 bf16_t;
typedef __bf16 v16bf __attribute__((ext_vector_type(16)));
typedef float  v8f   __attribute__((ext_vector_type(8)));

static constexpr int BB  = 128;   // batch
static constexpr int NN  = 402;   // nodes
static constexpr int FI  = 6;     // raw features
static constexpr int HH  = 128;   // hidden
static constexpr int CC  = 3;     // edge categories
static constexpr int NP  = 416;   // nodes padded to multiple of 32
static constexpr int MT  = NP / 16;   // 26 row tiles
static constexpr int NKC = NP / 32;   // 13 k chunks over nodes

// workspace layout (bytes)
static constexpr size_t SZ_XBF   = (size_t)BB * NP * HH * 2;      // current X, bf16 [b][n][h]
static constexpr size_t SZ_XWT   = (size_t)BB * CC * HH * NP * 2; // projected msgs, bf16 [b][c][h][n]
static constexpr size_t SZ_SELFT = (size_t)BB * HH * NP * 4;      // self term, f32 [b][h][n]
static constexpr size_t SZ_ENC   = (size_t)BB * NN * HH * 4;      // final enc_h, f32 [b][n][h]
static constexpr size_t SZ_DEGI  = (size_t)BB * CC * NP * 4;      // 1/deg, f32
static constexpr size_t SZ_WT    = (size_t)4 * HH * HH * 2;       // bf16 transposed weights

static constexpr size_t OFF_XBF   = 0;
static constexpr size_t OFF_XWT   = OFF_XBF   + SZ_XBF;
static constexpr size_t OFF_SELFT = OFF_XWT   + SZ_XWT;
static constexpr size_t OFF_ENC   = OFF_SELFT + SZ_SELFT;
static constexpr size_t OFF_DEGI  = OFF_ENC   + SZ_ENC;
static constexpr size_t OFF_WT    = OFF_DEGI  + SZ_DEGI;

union FragBF { v16bf v; uint4 q[2]; };
union Pack8BF { bf16_t h[8]; uint4 q; };

// --- 1. X0 = node_features @ W_embed, zero-padded to NP rows, bf16 ----------
__global__ void embed_kernel(const float* __restrict__ nf,
                             const float* __restrict__ we,
                             bf16_t* __restrict__ xbf) {
  int idx = blockIdx.x * 256 + threadIdx.x;
  if (idx >= BB * NP * HH) return;
  int h = idx & (HH - 1);
  int n = (idx >> 7) % NP;
  int b = idx / (NP * HH);
  float v = 0.f;
  if (n < NN) {
    const float* r = nf + ((size_t)b * NN + n) * FI;
#pragma unroll
    for (int f = 0; f < FI; ++f) v += r[f] * we[f * HH + h];
  }
  xbf[((size_t)b * NP + n) * HH + h] = (bf16_t)v;
}

// --- 2. deginv[b][c][i] = 1/max(deg,1); wave per row ------------------------
__global__ void deginv_kernel(const float* __restrict__ adj,
                              float* __restrict__ degi) {
  int wid  = (blockIdx.x * blockDim.x + threadIdx.x) >> 5;
  int lane = threadIdx.x & 31;
  if (wid >= BB * CC * NP) return;
  int i = wid % NP;
  int c = (wid / NP) % CC;
  int b = wid / (NP * CC);
  float s = 0.f;
  if (i < NN) {
    const float* row = adj + (((size_t)b * CC + c) * NN + i) * NN;
    for (int j = lane; j < NN; j += 32) s += row[j];
  }
#pragma unroll
  for (int o = 16; o > 0; o >>= 1) s += __shfl_xor(s, o, 32);
  if (lane == 0) degi[wid] = (i < NN) ? (1.0f / fmaxf(s, 1.0f)) : 1.0f;
}

// --- 3. transpose+convert weights: WT[w][g][f] (w<3: edge, w=3: self) -------
__global__ void wconv_kernel(const float* __restrict__ wl,   // [C][H][H]
                             const float* __restrict__ wsl,  // [H][H]
                             bf16_t* __restrict__ wt) {
  int idx = blockIdx.x * 256 + threadIdx.x;
  if (idx >= 4 * HH * HH) return;
  int w = idx / (HH * HH);
  int r = idx % (HH * HH);
  int g = r / HH, f = r % HH;
  float v = (w < 3) ? wl[(w * HH + f) * HH + g] : wsl[f * HH + g];
  wt[(w * HH + g) * HH + f] = (bf16_t)v;
}

// --- 4. projection: xwT[c] = X @ W_c (bf16), selfT = X @ Ws (f32) -----------
// grid(MT, B), block 256 (8 waves). 4 independent accumulator chains (w).
__global__ void proj_kernel(const bf16_t* __restrict__ xbf,
                            const bf16_t* __restrict__ wt,
                            bf16_t* __restrict__ xwt,
                            float* __restrict__ selft) {
  __shared__ bf16_t atile[16 * HH];   // 4 KB: 16 rows x 128 K
  const int mt = blockIdx.x, b = blockIdx.y;
  const int wave = threadIdx.x >> 5, lane = threadIdx.x & 31;
  const int ln = lane & 15;
  const bool hi = lane >= 16;
  const int kbA = hi ? 8 : 0, kbB = hi ? 16 : 0, off = hi ? 8 : 0;

  // cooperative coalesced load of 16 consecutive X rows (contiguous 4 KB)
  const uint4* src = (const uint4*)(xbf + ((size_t)b * NP + mt * 16) * HH);
  ((uint4*)atile)[threadIdx.x] = src[threadIdx.x];
  __syncthreads();

  const int nabs = wave * 16 + ln;     // absolute output column (h)
#pragma unroll
  for (int w = 0; w < 4; ++w) {
    v8f acc = {0.f, 0.f, 0.f, 0.f, 0.f, 0.f, 0.f, 0.f};
#pragma unroll
    for (int kc = 0; kc < HH / 32; ++kc) {
      FragBF a, bmat;
      const bf16_t* ap = atile + ln * HH + kc * 32;
      a.q[0] = *(const uint4*)(ap + kbB);        // K = kbB..kbB+7
      a.q[1] = *(const uint4*)(ap + kbB + 8);    // K = kbB+8..kbB+15
      const bf16_t* bp = wt + ((size_t)w * HH + nabs) * HH + kc * 32 + kbB;
      bmat.q[0] = ((const uint4*)bp)[0];
      bmat.q[1] = ((const uint4*)bp)[1];
      acc = __builtin_amdgcn_wmma_f32_16x16x32_bf16(
          false, a.v, false, bmat.v, (short)0, acc, false, false);
    }
    const int m0 = mt * 16 + off;   // first of 8 consecutive node rows
    if (w < 3) {
      Pack8BF o;
#pragma unroll
      for (int v = 0; v < 8; ++v) o.h[v] = (bf16_t)acc[v];
      *(uint4*)(xwt + (((size_t)b * CC + w) * HH + nabs) * NP + m0) = o.q;
    } else {
      float* sp = selft + ((size_t)b * HH + nabs) * NP + m0;
#pragma unroll
      for (int v = 0; v < 8; ++v) sp[v] = acc[v];
    }
  }
}

// --- 5. aggregate: msg = sum_c deginv_c/C * (adj_c @ xw_c); relu mix --------
// grid(MT, B), block 256. Whole 3 x 16 x 416 bf16 A-block staged in LDS once
// (single barrier), then 13 K-chunks x 3 independent WMMA chains per wave.
__global__ void agg_kernel(const float* __restrict__ adj,
                           const bf16_t* __restrict__ xwt,
                           const float* __restrict__ selft,
                           const float* __restrict__ degi,
                           bf16_t* __restrict__ xbf_next,
                           float* __restrict__ enc,
                           int last) {
  __shared__ bf16_t atile[CC * 16 * NP];   // 39,936 B
  const int mt = blockIdx.x, b = blockIdx.y;
  const int wave = threadIdx.x >> 5, lane = threadIdx.x & 31;
  const int ln = lane & 15;
  const bool hi = lane >= 16;
  const int kbA = hi ? 8 : 0, kbB = hi ? 16 : 0, off = hi ? 8 : 0;
  const int nabs = wave * 16 + ln;
  const int m0 = mt * 16 + off;

  // ---- stage: masked f32 -> bf16 conversion of adj rows, all edge types ----
  for (int c = 0; c < CC; ++c) {
    for (int i = 0; i < 16; ++i) {
      const int gi = mt * 16 + i;
      const float* row = adj + (((size_t)b * CC + c) * NN + gi) * NN;
      for (int j = threadIdx.x; j < NP; j += 256) {     // coalesced f32 reads
        float v = (gi < NN && j < NN) ? row[j] : 0.0f;
        atile[(c * 16 + i) * NP + j] = (bf16_t)v;
      }
    }
  }
  __syncthreads();

  // ---- 3 independent accumulator chains over 13 K-chunks ----
  v8f acc[CC];
#pragma unroll
  for (int c = 0; c < CC; ++c) acc[c] = (v8f){0.f, 0.f, 0.f, 0.f, 0.f, 0.f, 0.f, 0.f};

  const bf16_t* bbase = xwt + (((size_t)b * CC) * HH + nabs) * NP + kbB;
  for (int kc = 0; kc < NKC; ++kc) {
#pragma unroll
    for (int c = 0; c < CC; ++c) {
      FragBF a, bmat;
      const bf16_t* ap = atile + (c * 16 + ln) * NP + kc * 32;
      a.q[0] = *(const uint4*)(ap + kbB);
      a.q[1] = *(const uint4*)(ap + kbB + 8);
      const bf16_t* bp = bbase + (size_t)c * HH * NP + kc * 32;
      bmat.q[0] = ((const uint4*)bp)[0];
      bmat.q[1] = ((const uint4*)bp)[1];
      acc[c] = __builtin_amdgcn_wmma_f32_16x16x32_bf16(
          false, a.v, false, bmat.v, (short)0, acc[c], false, false);
    }
  }

  // ---- degree scaling, alpha mix, relu, store ----
  v8f msg = {0.f, 0.f, 0.f, 0.f, 0.f, 0.f, 0.f, 0.f};
#pragma unroll
  for (int c = 0; c < CC; ++c) {
    const float* dg = degi + ((size_t)b * CC + c) * NP + m0;
#pragma unroll
    for (int v = 0; v < 8; ++v) msg[v] += acc[c][v] * dg[v] * (1.0f / CC);
  }

  const float* sp = selft + ((size_t)b * HH + nabs) * NP + m0;
#pragma unroll
  for (int v = 0; v < 8; ++v) {
    float x = fmaxf(0.0f, 0.5f * sp[v] + 0.5f * msg[v]);   // ALPHA = 0.5
    int node = m0 + v;
    xbf_next[((size_t)b * NP + node) * HH + nabs] = (bf16_t)x;  // pads stay 0
    if (last && node < NN) enc[((size_t)b * NN + node) * HH + nabs] = x;
  }
}

// --- 6. trim enc_h[:, :-2] into output --------------------------------------
__global__ void trim_kernel(const float* __restrict__ enc,
                            float* __restrict__ out) {
  int idx = blockIdx.x * 256 + threadIdx.x;
  if (idx >= BB * 400 * HH) return;
  int h = idx & (HH - 1);
  int n = (idx >> 7) % 400;
  int b = idx / (400 * HH);
  out[idx] = enc[((size_t)b * NN + n) * HH + h];
}

// --- 7. mean / max / unbiased-std readout -----------------------------------
__global__ void readout_kernel(const float* __restrict__ enc,
                               float* __restrict__ out) {
  int b = blockIdx.x, h = threadIdx.x;
  float s = 0.f, ss = 0.f, mx = -3.4e38f;
  const float* p = enc + (size_t)b * NN * HH + h;
  for (int n = 0; n < NN; ++n) {
    float x = p[(size_t)n * HH];   // coalesced across h
    s += x; ss += x * x; mx = fmaxf(mx, x);
  }
  float mean = s / NN;
  float var = (ss - NN * mean * mean) / (NN - 1);   // ddof = 1
  out[b * 3 * HH + h]          = mean;
  out[b * 3 * HH + HH + h]     = mx;
  out[b * 3 * HH + 2 * HH + h] = sqrtf(fmaxf(var, 0.0f));
}

extern "C" void kernel_launch(void* const* d_in, const int* in_sizes, int n_in,
                              void* d_out, int out_size, void* d_ws, size_t ws_size,
                              hipStream_t stream) {
  const float* nf  = (const float*)d_in[0];   // [B,N,6]
  const float* adj = (const float*)d_in[1];   // [B,C,N,N]
  const float* we  = (const float*)d_in[2];   // [6,H]
  const float* w1  = (const float*)d_in[3];   // [L,C,H,H]
  const float* ws1 = (const float*)d_in[4];   // [L,H,H]
  const float* w2  = (const float*)d_in[5];
  const float* ws2 = (const float*)d_in[6];

  char* ws = (char*)d_ws;
  bf16_t* xbf   = (bf16_t*)(ws + OFF_XBF);
  bf16_t* xwt   = (bf16_t*)(ws + OFF_XWT);
  float*  selft = (float*)(ws + OFF_SELFT);
  float*  enc   = (float*)(ws + OFF_ENC);
  float*  degi  = (float*)(ws + OFF_DEGI);
  bf16_t* wt    = (bf16_t*)(ws + OFF_WT);

  float* out_h   = (float*)d_out;                       // [B, 3H]
  float* out_enc = out_h + (size_t)BB * 3 * HH;         // [B, 400, H]
  float* out_adj = out_enc + (size_t)BB * 400 * HH;     // [B, C, N, N]

  // adjacency passthrough output
  hipMemcpyAsync(out_adj, adj, (size_t)BB * CC * NN * NN * sizeof(float),
                 hipMemcpyDeviceToDevice, stream);

  embed_kernel<<<(BB * NP * HH + 255) / 256, 256, 0, stream>>>(nf, we, xbf);
  deginv_kernel<<<(BB * CC * NP) / 8, 256, 0, stream>>>(adj, degi);

  dim3 tiles(MT, BB);
  const float* Wstk[2]  = {w1, w2};
  const float* Wsstk[2] = {ws1, ws2};
  for (int stack = 0; stack < 2; ++stack) {
    for (int l = 0; l < 2; ++l) {
      const float* wl  = Wstk[stack]  + (size_t)l * CC * HH * HH;
      const float* wsl = Wsstk[stack] + (size_t)l * HH * HH;
      int last = (stack == 1 && l == 1) ? 1 : 0;
      wconv_kernel<<<(4 * HH * HH) / 256, 256, 0, stream>>>(wl, wsl, wt);
      proj_kernel<<<tiles, 256, 0, stream>>>(xbf, wt, xwt, selft);
      agg_kernel<<<tiles, 256, 0, stream>>>(adj, xwt, selft, degi, xbf, enc, last);
    }
  }

  trim_kernel<<<(BB * 400 * HH + 255) / 256, 256, 0, stream>>>(enc, out_enc);
  readout_kernel<<<BB, HH, 0, stream>>>(enc, out_h);
}
